// DistAttention_37108517437932
// MI455X (gfx1250) — matrix operand order repaired and checked by reading, not compiled, not run on hardware
//
#include <hip/hip_runtime.h>
#include <hip/hip_bf16.h>
#include <math.h>
#include <stdint.h>

typedef float  v8f  __attribute__((ext_vector_type(8)));
typedef float  v2f  __attribute__((ext_vector_type(2)));
typedef __bf16 v16bf __attribute__((ext_vector_type(16)));
typedef __bf16 v8bf  __attribute__((ext_vector_type(8)));
typedef unsigned int u32x4 __attribute__((ext_vector_type(4)));
typedef int i32x8 __attribute__((ext_vector_type(8)));
typedef int i32x4 __attribute__((ext_vector_type(4)));

constexpr int NL  = 256;   // sequence length
constexpr int ND  = 1024;  // model dim
constexpr int NH  = 16;    // heads
constexpr int NDH = 64;    // head dim

__device__ __forceinline__ v16bf mk16(v8bf lo, v8bf hi) {
  v16bf r;
#pragma unroll
  for (int i = 0; i < 8; ++i) { r[i] = lo[i]; r[i + 8] = hi[i]; }
  return r;
}

__device__ __forceinline__ float softplus_f(float x) {
  return (x > 20.0f) ? x : log1pf(__expf(x));
}

// ---------------------------------------------------------------------------
// Convert 8 weight matrices W[k][n] (f32, row-major, x@W) into bf16 WT[n][k]
// so WMMA B-fragments are contiguous-K vector loads.
// ---------------------------------------------------------------------------
__global__ void k_convert_weights(const float* __restrict__ w0, const float* __restrict__ w1,
                                  const float* __restrict__ w2, const float* __restrict__ w3,
                                  const float* __restrict__ w4, const float* __restrict__ w5,
                                  const float* __restrict__ w6, const float* __restrict__ w7,
                                  __bf16* __restrict__ wt) {
  unsigned idx = blockIdx.x * blockDim.x + threadIdx.x;   // 8 * 1M
  unsigned w   = idx >> 20;
  unsigned rem = idx & ((1u << 20) - 1);
  unsigned n   = rem >> 10;
  unsigned k   = rem & 1023u;
  const float* W;
  switch (w) {
    case 0: W = w0; break; case 1: W = w1; break;
    case 2: W = w2; break; case 3: W = w3; break;
    case 4: W = w4; break; case 5: W = w5; break;
    case 6: W = w6; break; default: W = w7; break;
  }
  wt[idx] = (__bf16)W[(size_t)k * ND + n];
}

__global__ void k_convert_act(const float* __restrict__ mu, const float* __restrict__ var,
                              __bf16* __restrict__ amu, __bf16* __restrict__ avar) {
  int idx = blockIdx.x * blockDim.x + threadIdx.x;        // NL*ND
  amu[idx]  = (__bf16)mu[idx];
  avar[idx] = (__bf16)var[idx];
}

// ---------------------------------------------------------------------------
// C[256x1024] = A[256x1024](bf16) @ W (via WT[n][k] bf16) + bias, optional
// softplus. grid=(16,8), block=256 (8 waves); wave -> one 16x16 N-tile,
// K-loop 1024 in steps of 32 with v_wmma_f32_16x16x32_bf16. The 32 KB A
// block-row is staged into LDS by the Tensor Data Mover (one descriptor,
// issued by wave 0, fenced with s_wait_tensorcnt before the block barrier).
// ---------------------------------------------------------------------------
__global__ void __launch_bounds__(256) k_gemm(const __bf16* __restrict__ A,
                                              const __bf16* __restrict__ WT,
                                              const float* __restrict__ bias,
                                              float* __restrict__ C, int act) {
  __shared__ __bf16 As[16 * ND];                          // 32 KB
  const int tid = threadIdx.x;
  const int m0  = blockIdx.x * 16;
  const int n0  = blockIdx.y * 128 + (tid >> 5) * 16;

  if ((tid >> 5) == 0) {  // TDM: 2-D tile 16 rows x 1024 cols of 2-byte data
    unsigned lds_off = (unsigned)(uintptr_t)(&As[0]);
    unsigned long long ga =
        (unsigned long long)(uintptr_t)(A + (size_t)m0 * ND);
    u32x4 g0;
    g0.x = 1u;                                            // count=1 (valid, user)
    g0.y = lds_off;                                       // lds_addr (bytes)
    g0.z = (unsigned)ga;                                  // global_addr[31:0]
    g0.w = (unsigned)((ga >> 32) & 0x01ffffffu) | (2u << 30);  // addr[56:32]|type=2
    i32x8 g1;
    g1[0] = 0x00010000;                                   // data_size=1 (2 bytes)
    g1[1] = (int)(1024u << 16);                           // tensor_dim0 = 1024
    g1[2] = (int)(256u << 16);                            // tensor_dim1 = 256
    g1[3] = (int)(1024u << 16);                           // tile_dim0 = 1024
    g1[4] = 16;                                           // tile_dim1 = 16
    g1[5] = 1024;                                         // tensor_dim0_stride
    g1[6] = 0;
    g1[7] = 0;
    i32x4 gz = {0, 0, 0, 0};
#if __clang_major__ >= 23
    i32x8 gz8 = {0, 0, 0, 0, 0, 0, 0, 0};
    __builtin_amdgcn_tensor_load_to_lds(g0, g1, gz, gz, gz8, 0);
#else
    __builtin_amdgcn_tensor_load_to_lds(g0, g1, gz, gz, 0);
#endif
    __builtin_amdgcn_s_wait_tensorcnt(0);
  }
  __syncthreads();

  const int lane = tid & 31;
  const int hl   = lane >> 4;
  const int r    = lane & 15;
  const __bf16* Brow = WT + (size_t)(n0 + r) * ND;

  v8f acc = {0.f, 0.f, 0.f, 0.f, 0.f, 0.f, 0.f, 0.f};
#pragma unroll 4
  for (int kk = 0; kk < ND; kk += 32) {
    __builtin_prefetch(&Brow[kk + 128], 0, 1);
    v8bf alo = *(const v8bf*)&As[r * ND + kk + hl * 8];        // A: K-chunks of 8, +16 apart
    v8bf ahi = *(const v8bf*)&As[r * ND + kk + hl * 8 + 16];
    v8bf blo = *(const v8bf*)&Brow[kk + hl * 8];               // B: K-chunks of 8, +16 apart
    v8bf bhi = *(const v8bf*)&Brow[kk + hl * 8 + 16];
    acc = __builtin_amdgcn_wmma_f32_16x16x32_bf16(
        false, mk16(alo, ahi), false, mk16(blo, bhi), (short)0, acc, false, false);
  }

  const float bn = bias[n0 + r];
#pragma unroll
  for (int i = 0; i < 8; ++i) {
    float v = acc[i] + bn;
    if (act) v = softplus_f(v);
    C[(size_t)(m0 + i + hl * 8) * ND + (n0 + r)] = v;
  }
}

// ---------------------------------------------------------------------------
// Per-(head,key) constant: c = sum_d ( Km^2/(2 Kv) + 0.5 log Kv - 0.5 )
// ---------------------------------------------------------------------------
__global__ void k_ck(const float* __restrict__ Km, const float* __restrict__ Kv,
                     float* __restrict__ c) {
  int idx = blockIdx.x * blockDim.x + threadIdx.x;        // NH*NL
  int h = idx >> 8, k = idx & 255;
  const float* kmr = Km + (size_t)k * ND + h * NDH;
  const float* kvr = Kv + (size_t)k * ND + h * NDH;
  float s = 0.f;
#pragma unroll 8
  for (int d = 0; d < NDH; ++d) {
    float kv = kvr[d], km = kmr[d];
    s += 0.5f * (km * km / kv + __logf(kv) - 1.0f);
  }
  c[idx] = s;
}

// VmT[n][k] = Vm[k][n] in bf16 (B operand for the P@V GEMM)
__global__ void k_transposeV(const float* __restrict__ Vm, const float* __restrict__ Vv,
                             __bf16* __restrict__ VmT, __bf16* __restrict__ VvT) {
  int idx = blockIdx.x * blockDim.x + threadIdx.x;        // ND*NL
  int n = idx >> 8, k = idx & 255;
  VmT[idx] = (__bf16)Vm[(size_t)k * ND + n];
  VvT[idx] = (__bf16)Vv[(size_t)k * ND + n];
}

// ---------------------------------------------------------------------------
// Fused attention: one block per (q-block of 16, head). 8 waves.
// Phase 1: KL logits via f32 WMMA (K=64, two GEMMs folded into one acc by
//          negating the second B operand). Phase 2: row softmax in LDS.
// Phase 3: out_mu = P@Vm, out_var = P^2@Vv via bf16 WMMA, written in
//          head-combined layout as bf16 for the final projections.
// ---------------------------------------------------------------------------
__global__ void __launch_bounds__(256) k_attn(
    const float* __restrict__ Qm, const float* __restrict__ Qv,
    const float* __restrict__ Km, const float* __restrict__ Kv,
    const float* __restrict__ Cc, const __bf16* __restrict__ VmT,
    const __bf16* __restrict__ VvT, __bf16* __restrict__ Omu,
    __bf16* __restrict__ Ovar) {
  __shared__ float  sc[16 * NL];      // 16 KB scores
  __shared__ __bf16 pb[16 * NL];      //  8 KB probs
  __shared__ __bf16 pb2[16 * NL];     //  8 KB probs^2
  __shared__ float  red[16 * 16];
  __shared__ float  rowmax[16];
  __shared__ float  rowsum[16];

  const int tid  = threadIdx.x;
  const int wave = tid >> 5;
  const int lane = tid & 31;
  const int hl   = lane >> 4;
  const int r    = lane & 15;
  const int q0   = blockIdx.x * 16;
  const int h0   = blockIdx.y * NDH;   // head column offset
  const int hc   = blockIdx.y * NL;    // c_k offset

  // ---- Phase 1: scores for 16 q-rows x 256 keys (wave -> 2 key-tiles) ----
  const float* qmr = Qm + (size_t)(q0 + r) * ND + h0;
  const float* qvr = Qv + (size_t)(q0 + r) * ND + h0;
#pragma unroll
  for (int t = 0; t < 2; ++t) {
    const int kt   = wave * 2 + t;
    const int krow = kt * 16 + r;
    const float* kmr = Km + (size_t)krow * ND + h0;
    const float* kvr = Kv + (size_t)krow * ND + h0;
    v8f acc = {0.f, 0.f, 0.f, 0.f, 0.f, 0.f, 0.f, 0.f};
#pragma unroll
    for (int ds = 0; ds < NDH; ds += 4) {
      const int d0 = ds + hl * 2;
      float qm0 = qmr[d0], qm1 = qmr[d0 + 1];
      float qv0 = qvr[d0], qv1 = qvr[d0 + 1];
      float kv0 = kvr[d0], kv1 = kvr[d0 + 1];
      float km0 = kmr[d0], km1 = kmr[d0 + 1];
      float i0 = 1.0f / kv0, i1 = 1.0f / kv1;
      v2f aG = {qm0 * qm0 + qv0, qm1 * qm1 + qv1};   // (Qm^2 + Qv)
      v2f bA = {0.5f * i0, 0.5f * i1};               // 1/(2 Kv)
      v2f aQ = {qm0, qm1};                           // Qm
      v2f bB = {-km0 * i0, -km1 * i1};               // -Km/Kv (fold subtraction)
      acc = __builtin_amdgcn_wmma_f32_16x16x4_f32(false, aG, false, bA,
                                                  (short)0, acc, false, false);
      acc = __builtin_amdgcn_wmma_f32_16x16x4_f32(false, aQ, false, bB,
                                                  (short)0, acc, false, false);
    }
    const float ck = Cc[hc + kt * 16 + r];
#pragma unroll
    for (int i = 0; i < 8; ++i)
      sc[(i + hl * 8) * NL + kt * 16 + r] = -0.125f * (acc[i] + ck);
  }
  __syncthreads();

  // ---- Phase 2: softmax over 16 rows of 256 ----
  {
    const int rr  = tid >> 4;
    const int seg = tid & 15;
    float* srow = sc + rr * NL + seg * 16;
    float mx = srow[0];
#pragma unroll
    for (int j = 1; j < 16; ++j) mx = fmaxf(mx, srow[j]);
    red[rr * 16 + seg] = mx;
    __syncthreads();
    if (seg == 0) {
      float m = red[rr * 16];
#pragma unroll
      for (int j = 1; j < 16; ++j) m = fmaxf(m, red[rr * 16 + j]);
      rowmax[rr] = m;
    }
    __syncthreads();
    const float m = rowmax[rr];
    float s = 0.f;
#pragma unroll
    for (int j = 0; j < 16; ++j) {
      float e = __expf(srow[j] - m);
      srow[j] = e;
      s += e;
    }
    red[rr * 16 + seg] = s;
    __syncthreads();
    if (seg == 0) {
      float t = 0.f;
#pragma unroll
      for (int j = 0; j < 16; ++j) t += red[rr * 16 + j];
      rowsum[rr] = t;
    }
    __syncthreads();
    const float inv = 1.0f / rowsum[rr];
#pragma unroll
    for (int j = 0; j < 16; ++j) {
      float p = srow[j] * inv;
      pb[rr * NL + seg * 16 + j]  = (__bf16)p;
      pb2[rr * NL + seg * 16 + j] = (__bf16)(p * p);
    }
  }
  __syncthreads();

  // ---- Phase 3: waves 0-3 -> P@Vm tile n, waves 4-7 -> P^2@Vv tile n ----
  {
    const int sel = wave >> 2;
    const int nt  = wave & 3;
    const __bf16* P  = sel ? pb2 : pb;
    const __bf16* BT = (sel ? VvT : VmT) + (size_t)(h0 + nt * 16 + r) * NL;
    v8f acc = {0.f, 0.f, 0.f, 0.f, 0.f, 0.f, 0.f, 0.f};
#pragma unroll
    for (int kk = 0; kk < NL; kk += 32) {
      v8bf alo = *(const v8bf*)&P[r * NL + kk + hl * 8];
      v8bf ahi = *(const v8bf*)&P[r * NL + kk + hl * 8 + 16];
      v8bf blo = *(const v8bf*)&BT[kk + hl * 8];
      v8bf bhi = *(const v8bf*)&BT[kk + hl * 8 + 16];
      acc = __builtin_amdgcn_wmma_f32_16x16x32_bf16(
          false, mk16(alo, ahi), false, mk16(blo, bhi), (short)0, acc, false, false);
    }
    __bf16* O = sel ? Ovar : Omu;
    const int col = h0 + nt * 16 + r;
#pragma unroll
    for (int i = 0; i < 8; ++i)
      O[(size_t)(q0 + i + hl * 8) * ND + col] = (__bf16)acc[i];
  }
}

// ---------------------------------------------------------------------------
extern "C" void kernel_launch(void* const* d_in, const int* in_sizes, int n_in,
                              void* d_out, int out_size, void* d_ws, size_t ws_size,
                              hipStream_t stream) {
  const float* mu  = (const float*)d_in[0];
  const float* var = (const float*)d_in[1];
  const float* w[8];
  const float* b[8];
  for (int i = 0; i < 8; ++i) {           // q_mu,q_var,k_mu,k_var,v_mu,v_var,o_mu,o_var
    w[i] = (const float*)d_in[2 + 2 * i];
    b[i] = (const float*)d_in[3 + 2 * i];
  }

  char* p = (char*)d_ws;                  // ~25 MB total, all 256B-multiple chunks
  __bf16* Amu  = (__bf16*)p; p += (size_t)NL * ND * 2;
  __bf16* Avar = (__bf16*)p; p += (size_t)NL * ND * 2;
  __bf16* WT   = (__bf16*)p; p += (size_t)8 * ND * ND * 2;
  float*  Qm   = (float*)p;  p += (size_t)NL * ND * 4;
  float*  Qv   = (float*)p;  p += (size_t)NL * ND * 4;
  float*  Km   = (float*)p;  p += (size_t)NL * ND * 4;
  float*  Kv   = (float*)p;  p += (size_t)NL * ND * 4;
  float*  Vm   = (float*)p;  p += (size_t)NL * ND * 4;
  float*  Vv   = (float*)p;  p += (size_t)NL * ND * 4;
  float*  Cc   = (float*)p;  p += (size_t)NH * NL * 4;
  __bf16* VmT  = (__bf16*)p; p += (size_t)ND * NL * 2;
  __bf16* VvT  = (__bf16*)p; p += (size_t)ND * NL * 2;
  __bf16* Omu  = (__bf16*)p; p += (size_t)NL * ND * 2;
  __bf16* Ovar = (__bf16*)p; p += (size_t)NL * ND * 2;

  float* out_mu  = (float*)d_out;
  float* out_var = out_mu + (size_t)NL * ND;

  k_convert_weights<<<(8u << 20) / 256, 256, 0, stream>>>(
      w[0], w[1], w[2], w[3], w[4], w[5], w[6], w[7], WT);
  k_convert_act<<<(NL * ND) / 256, 256, 0, stream>>>(mu, var, Amu, Avar);

  dim3 gg(16, 8);
  k_gemm<<<gg, 256, 0, stream>>>(Amu,  WT + (size_t)0 * ND * ND, b[0], Qm, 0);
  k_gemm<<<gg, 256, 0, stream>>>(Avar, WT + (size_t)1 * ND * ND, b[1], Qv, 1);
  k_gemm<<<gg, 256, 0, stream>>>(Amu,  WT + (size_t)2 * ND * ND, b[2], Km, 0);
  k_gemm<<<gg, 256, 0, stream>>>(Avar, WT + (size_t)3 * ND * ND, b[3], Kv, 1);
  k_gemm<<<gg, 256, 0, stream>>>(Amu,  WT + (size_t)4 * ND * ND, b[4], Vm, 0);
  k_gemm<<<gg, 256, 0, stream>>>(Avar, WT + (size_t)5 * ND * ND, b[5], Vv, 1);

  k_ck<<<(NH * NL) / 256, 256, 0, stream>>>(Km, Kv, Cc);
  k_transposeV<<<(ND * NL) / 256, 256, 0, stream>>>(Vm, Vv, VmT, VvT);

  k_attn<<<dim3(16, 16), 256, 0, stream>>>(Qm, Qv, Km, Kv, Cc, VmT, VvT, Omu, Ovar);

  k_gemm<<<gg, 256, 0, stream>>>(Omu,  WT + (size_t)6 * ND * ND, b[6], out_mu, 0);
  k_gemm<<<gg, 256, 0, stream>>>(Ovar, WT + (size_t)7 * ND * ND, b[7], out_var, 1);
}
